// ComplexDenoisingAutoencoder_10256381903146
// MI455X (gfx1250) — matrix operand order repaired and checked by reading, not compiled, not run on hardware
//
#include <hip/hip_runtime.h>

// CDNA5 / gfx1250 fused per-patch denoising autoencoder.
// One workgroup (8 wave32) per (patch-index, image) pair; all intermediates in LDS;
// convs as implicit GEMMs on v_wmma_f32_16x16x32_f16.
//
// K packing uses 12 slots per input channel (9 real taps + 3 dummies). Dummy slots
// read window row 3 (kept in bounds by the 69-row plane), and the corresponding
// weight-fragment entries are zero, so the A-gather needs no masking at all.

typedef __attribute__((ext_vector_type(16))) _Float16 v16h;
typedef __attribute__((ext_vector_type(8)))  float    v8f;

namespace {

constexpr int IMG      = 256;
constexpr int PATCHSZ  = 64;
constexpr int NP       = 16;
constexpr int BATCH    = 64;
constexpr int CIN      = 2;
constexpr int HID      = 8;
constexpr int PADW     = 66;                  // 64 + 1-halo each side
constexpr int PADH     = 69;                  // 66 rows + 3 always-zero dummy rows
constexpr int PLANE    = PADH * PADW;         // 4554 halfs per padded channel plane
constexpr int BUF_HALFS = 8 * PLANE;          // 36432 halfs per ping/pong buffer
constexpr int NPOS     = PATCHSZ * PATCHSZ;   // 4096 spatial positions per patch
constexpr int KPC      = 12;                  // K slots per input channel (9 real + 3 dummy)

// One conv layer as implicit GEMM:  D[M=4096 pos, N=O ch] = A[M,K=I*12] x B[K,O] + bias
template<int I, int O, int NCHUNK, bool RELU, bool LAST>
__device__ __forceinline__ void conv_layer(
        const _Float16* __restrict__ inbuf,   // padded f16 input planes
        _Float16* __restrict__ outbuf,        // padded f16 output planes (!LAST)
        float* __restrict__ stage,            // f32 staging (LAST)
        const v16h (&bw)[NCHUNK],             // weight B-fragments (per-lane)
        v8f cinit,                            // bias-seeded C fragment
        int wave, int mlo, int lhalf)
{
    // Per-lane im2col offsets: A-fragment element e of chunk c holds
    // K = c*32 + e + 16*lhalf   (contiguous order, matching the B fragments).
    // K -> (ci = K/12, r = K%12); r in [9,12) are dummy slots (zero weights) reading window row 3.
    int off[NCHUNK][16];
#pragma unroll
    for (int c = 0; c < NCHUNK; ++c) {
#pragma unroll
        for (int e = 0; e < 16; ++e) {
            int kk = c * 32 + e + 16 * lhalf;
            int ci = kk / KPC, r = kk % KPC;
            off[c][e] = ci * PLANE + (r / 3) * PADW + (r % 3);
        }
    }
    constexpr int TILES_PER_WAVE = NPOS / 16 / 8;   // 32
    for (int j = 0; j < TILES_PER_WAVE; ++j) {
        const int t  = wave * TILES_PER_WAVE + j;
        const int m0 = t * 16 + mlo;                       // this lane's A row
        const int rowbase = (m0 >> 6) * PADW + (m0 & 63);  // window top-left in padded plane
        v8f acc = cinit;
#pragma unroll
        for (int c = 0; c < NCHUNK; ++c) {
            v16h a;
#pragma unroll
            for (int e = 0; e < 16; ++e)
                a[e] = inbuf[rowbase + off[c][e]];         // unconditional gather
            acc = __builtin_amdgcn_wmma_f32_16x16x32_f16(
                    false, a, false, bw[c], (short)0, acc, false, false);
        }
        // D layout: lane covers column n = mlo; VGPR r holds row m = t*16 + r + 8*lhalf
        if (mlo < O) {
#pragma unroll
            for (int r = 0; r < 8; ++r) {
                const int m = t * 16 + r + 8 * lhalf;
                float v = acc[r];
                if (RELU) v = v > 0.0f ? v : 0.0f;
                if (LAST) {
                    stage[mlo * NPOS + m] = v;
                } else {
                    outbuf[mlo * PLANE + ((m >> 6) + 1) * PADW + (m & 63) + 1] = (_Float16)v;
                }
            }
        }
    }
}

__global__ __launch_bounds__(256) void cae_fused_kernel(
    const float* __restrict__ x,
    const float* __restrict__ w1, const float* __restrict__ b1,
    const float* __restrict__ w2, const float* __restrict__ b2,
    const float* __restrict__ w3, const float* __restrict__ b3,
    float* __restrict__ out)
{
    __shared__ _Float16 lds[2 * BUF_HALFS];                   // ~142 KB of the 320 KB WGP LDS
    _Float16* buf0  = lds;
    _Float16* buf1  = lds + BUF_HALFS;
    float*    stage = reinterpret_cast<float*>(lds + BUF_HALFS); // aliases buf1 (free at layer 3)

    const int tid   = threadIdx.x;
    const int lane  = tid & 31;
    const int lhalf = lane >> 4;
    const int mlo   = lane & 15;
    const int wave  = tid >> 5;
    const int wg    = blockIdx.x;
    const int img   = wg >> 4;          // image b
    const int pi    = wg & 15;          // patch index i = py*4 + px
    const int py    = pi >> 2, px = pi & 3;

    const float* xpatch = x + (((size_t)img * CIN) * IMG + py * PATCHSZ) * IMG + px * PATCHSZ;
    __builtin_prefetch(xpatch, 0, 1);   // global_prefetch_b8: warm the patch while we build fragments

    // ---- constant weight (B) fragments + bias-seeded C fragments, built once ----
    // B 32x16 f16 layout: lane holds column n = lane&15; element e -> K = e + 16*lhalf.
    const float* W1 = w1 + pi * HID * CIN * 9;
    const float* W2 = w2 + pi * HID * HID * 9;
    const float* W3 = w3 + pi * CIN * HID * 9;
    v16h bw1[1], bw2[3], bw3[3];
#pragma unroll
    for (int e = 0; e < 16; ++e) {
        int  kk = e + 16 * lhalf;
        int  ci = kk / KPC, r = kk % KPC;
        bool v  = (ci < CIN) && (r < 9);
        int  cic = v ? ci : 0, rc = v ? r : 0;
        bw1[0][e] = (mlo < HID && v)
                  ? (_Float16)W1[(mlo * CIN + cic) * 9 + rc] : (_Float16)0.0f;
    }
#pragma unroll
    for (int c = 0; c < 3; ++c) {
#pragma unroll
        for (int e = 0; e < 16; ++e) {
            int  kk = c * 32 + e + 16 * lhalf;
            int  ci = kk / KPC, r = kk % KPC;
            bool v  = (ci < HID) && (r < 9);
            int  cic = v ? ci : 0, rc = v ? r : 0;
            bw2[c][e] = (mlo < HID && v)
                      ? (_Float16)W2[(mlo * HID + cic) * 9 + rc] : (_Float16)0.0f;
            bw3[c][e] = (mlo < CIN && v)
                      ? (_Float16)W3[(mlo * HID + cic) * 9 + rc] : (_Float16)0.0f;
        }
    }
    v8f c1, c2, c3;
    {
        float v1 = (mlo < HID) ? b1[pi * HID + mlo] : 0.0f;
        float v2 = (mlo < HID) ? b2[pi * HID + mlo] : 0.0f;
        float v3 = (mlo < CIN) ? b3[pi * CIN + mlo] : 0.0f;
#pragma unroll
        for (int r = 0; r < 8; ++r) { c1[r] = v1; c2[r] = v2; c3[r] = v3; }
    }

    // ---- zero both LDS buffers (provides conv halo + dummy-slot zero rows) ----
    {
        uint32_t* z = reinterpret_cast<uint32_t*>(lds);
        constexpr int NDW = (2 * BUF_HALFS) / 2;
        for (int k2 = tid; k2 < NDW; k2 += 256) z[k2] = 0u;
    }
    __syncthreads();

    // ---- load input patch, f32 -> f16, into buf0 interior (coalesced float4 rows) ----
#pragma unroll
    for (int k2 = 0; k2 < 8; ++k2) {
        int idx = tid + k2 * 256;            // float4 index, 0..2047
        int c   = idx >> 10;                 // channel
        int rm  = idx & 1023;
        int yy  = rm >> 4;
        int xx  = (rm & 15) << 2;
        float4 v = *reinterpret_cast<const float4*>(xpatch + ((size_t)c * IMG + yy) * IMG + xx);
        _Float16* d = buf0 + c * PLANE + (yy + 1) * PADW + (xx + 1);
        d[0] = (_Float16)v.x; d[1] = (_Float16)v.y;
        d[2] = (_Float16)v.z; d[3] = (_Float16)v.w;
    }
    __syncthreads();

    // ---- the 3 conv layers, all in LDS ----
    conv_layer<CIN, HID, 1, true,  false>(buf0, buf1, nullptr, bw1, c1, wave, mlo, lhalf);
    __syncthreads();
    conv_layer<HID, HID, 3, true,  false>(buf1, buf0, nullptr, bw2, c2, wave, mlo, lhalf);
    __syncthreads();
    conv_layer<HID, CIN, 3, false, true >(buf0, nullptr, stage, bw3, c3, wave, mlo, lhalf);
    __syncthreads();

    // ---- coalesced float4 writeback to both output layouts ----
    float* outc = out;                                       // combined [B,C,256,256]
    float* outd = out + (size_t)BATCH * CIN * IMG * IMG;     // denoised [(i*B+b),C,64,64]
#pragma unroll
    for (int k2 = 0; k2 < 8; ++k2) {
        int idx = tid + k2 * 256;
        int c   = idx >> 10;
        int rm  = idx & 1023;
        int yy  = rm >> 4;
        int xx  = (rm & 15) << 2;
        float4 v = *reinterpret_cast<const float4*>(stage + c * NPOS + yy * PATCHSZ + xx);
        *reinterpret_cast<float4*>(outc +
            (((size_t)img * CIN + c) * IMG + py * PATCHSZ + yy) * IMG + px * PATCHSZ + xx) = v;
        *reinterpret_cast<float4*>(outd +
            ((size_t)(pi * BATCH + img) * CIN + c) * NPOS + yy * PATCHSZ + xx) = v;
    }
}

} // namespace

extern "C" void kernel_launch(void* const* d_in, const int* in_sizes, int n_in,
                              void* d_out, int out_size, void* d_ws, size_t ws_size,
                              hipStream_t stream) {
    (void)in_sizes; (void)n_in; (void)out_size; (void)d_ws; (void)ws_size;
    const float* x  = (const float*)d_in[0];
    const float* w1 = (const float*)d_in[1];
    const float* b1 = (const float*)d_in[2];
    const float* w2 = (const float*)d_in[3];
    const float* b2 = (const float*)d_in[4];
    const float* w3 = (const float*)d_in[5];
    const float* b3 = (const float*)d_in[6];
    float* out = (float*)d_out;
    cae_fused_kernel<<<dim3(NP * BATCH), dim3(256), 0, stream>>>(x, w1, b1, w2, b2, w3, b3, out);
}
